// BipolarLinearAttention_65996467471031
// MI455X (gfx1250) — matrix operand order repaired and checked by reading, not compiled, not run on hardware
//
#include <hip/hip_runtime.h>

typedef __bf16 bf16;
typedef __bf16 v16bf __attribute__((ext_vector_type(16)));
typedef __bf16 v8bf  __attribute__((ext_vector_type(8)));
typedef float  v8f   __attribute__((ext_vector_type(8)));

#define B_      8
#define C_      256
#define N_      4096
#define HEADS_  8
#define HD_     32
#define BETA_   0.5f
#define THR_    1.0f
#define ELU0_   0.36787944117144233f   /* elu(-1)+1 = exp(-1) */
#define SCALE_  0.17677669529663687f   /* 1/sqrt(32) */

#define CAT16(lo, hi) __builtin_shufflevector(lo, hi, 0,1,2,3,4,5,6,7,8,9,10,11,12,13,14,15)

static __device__ __forceinline__ v8f wmma_bf16(v16bf a, v16bf b, v8f c) {
  return __builtin_amdgcn_wmma_f32_16x16x32_bf16(false, a, false, b, (short)0, c, false, false);
}

// pack two f32 -> packed bf16 pair (u32)
static __device__ __forceinline__ unsigned pkbf(float a, float b) {
  union { bf16 h[2]; unsigned u; } p;
  p.h[0] = (bf16)a; p.h[1] = (bf16)b;
  return p.u;
}

// ---- CDNA5 async global->LDS copies (ASYNCcnt-tracked) ----------------------
static __device__ __forceinline__ void async1_b128(unsigned lds, const void* g) {
  asm volatile("global_load_async_to_lds_b128 %0, %1, off"
               :: "v"(lds), "v"((unsigned long long)(size_t)g) : "memory");
}
static __device__ __forceinline__ void async2_b128(unsigned lds, const void* g) {
  asm volatile("global_load_async_to_lds_b128 %0, %1, off\n\t"
               "global_load_async_to_lds_b128 %0, %1, off offset:16"
               :: "v"(lds), "v"((unsigned long long)(size_t)g) : "memory");
}
static __device__ __forceinline__ void async4_b128(unsigned lds, const void* g) {
  asm volatile("global_load_async_to_lds_b128 %0, %1, off\n\t"
               "global_load_async_to_lds_b128 %0, %1, off offset:16\n\t"
               "global_load_async_to_lds_b128 %0, %1, off offset:32\n\t"
               "global_load_async_to_lds_b128 %0, %1, off offset:48"
               :: "v"(lds), "v"((unsigned long long)(size_t)g) : "memory");
}
static __device__ __forceinline__ void wait_async0() {
  asm volatile("s_wait_asynccnt 0x0" ::: "memory");
}

// Fragment-packed layout helper: [group nt][lane l 0..15][kh 0..1][kl 0..15]
// element offset = ((nt*16 + l)*2 + kh)*16 + kl     (each lane run = 32B)

// ---------------------------------------------------------------------------
// Kernel 1: qkv conv1x1 GEMM (bf16 WMMA, f32 acc) + fused LIF.
// grid = (N/256, C/64, 3*B); block 256 (8 waves, 8 WMMA each per K-step).
// q spikes -> fragment-packed global layout; k/v spikes -> channel-major.
// ---------------------------------------------------------------------------
__global__ __launch_bounds__(256)
void qkv_gemm_lif(const float* __restrict__ x,
                  const float* __restrict__ Wq, const float* __restrict__ Wk,
                  const float* __restrict__ Wv,
                  const float* __restrict__ qmem, const float* __restrict__ kmem,
                  const float* __restrict__ vmem,
                  float* __restrict__ out_base,
                  bf16* __restrict__ qP, bf16* __restrict__ k_ws,
                  bf16* __restrict__ v_ws)
{
  const int b   = blockIdx.z % B_;
  const int mat = blockIdx.z / B_;            // 0=q, 1=k, 2=v
  const int m0  = blockIdx.y * 64;
  const int n0  = blockIdx.x * 256;

  const float* W      = (mat == 0) ? Wq : (mat == 1) ? Wk : Wv;
  const float* memin  = (mat == 0) ? qmem : (mat == 1) ? kmem : vmem;
  float*       memout = out_base + (size_t)(1 + mat) * B_ * C_ * N_;

  __shared__ bf16 sW[64][48];                 // row stride 96B (16B aligned)
  __shared__ bf16 sXf[16 * 16 * 2 * 16];      // fragment-packed, 16KB

  const int t = threadIdx.x, wave = t >> 5, lane = t & 31;
  const int rT = wave & 3, cG = wave >> 2;
  const int l15 = lane & 15, hiL = lane >> 4;

  v8f acc[8];
  #pragma unroll
  for (int i = 0; i < 8; ++i) acc[i] = v8f{};

  const int wrow = t >> 2, wcol = (t & 3) * 8;     // W staging
  const int xrq  = t >> 5;                         // X staging: 4-row quad
  const int xc0  = (t & 31) * 8;                   // 8 cols per thread

  for (int k0 = 0; k0 < C_; k0 += 32) {
    { // W tile 64x32: f32 -> bf16, one b128 LDS store
      const float4* s4 = (const float4*)(W + (size_t)(m0 + wrow) * C_ + k0 + wcol);
      float4 f0 = s4[0], f1 = s4[1];
      uint4 u;
      u.x = pkbf(f0.x, f0.y); u.y = pkbf(f0.z, f0.w);
      u.z = pkbf(f1.x, f1.y); u.w = pkbf(f1.z, f1.w);
      *(uint4*)&sW[wrow][wcol] = u;
    }
    { // X tile 32x256: 4 k-rows x 8 cols per thread, packed b64 stores
      float4 f[4][2];
      #pragma unroll
      for (int r = 0; r < 4; ++r) {
        const float4* s4 =
            (const float4*)(x + ((size_t)b * C_ + (k0 + 4 * xrq + r)) * N_ + n0 + xc0);
        f[r][0] = s4[0]; f[r][1] = s4[1];
      }
      const float* p0 = (const float*)&f[0][0];
      const float* p1 = (const float*)&f[1][0];
      const float* p2 = (const float*)&f[2][0];
      const float* p3 = (const float*)&f[3][0];
      const int kh = (4 * xrq) >> 4, kl0 = (4 * xrq) & 15;
      #pragma unroll
      for (int j = 0; j < 8; ++j) {
        int col = xc0 + j;
        unsigned off = (unsigned)((((col >> 4) * 16 + (col & 15)) * 2 + kh) * 16 + kl0);
        uint2 u2; u2.x = pkbf(p0[j], p1[j]); u2.y = pkbf(p2[j], p3[j]);
        *(uint2*)(sXf + off) = u2;
      }
    }
    __syncthreads();

    const int m  = rT * 16 + l15;
    const int kb = hiL ? 16 : 0;
    v8bf alo = *(const v8bf*)&sW[m][kb];
    v8bf ahi = *(const v8bf*)&sW[m][kb + 8];
    v16bf a = CAT16(alo, ahi);

    #pragma unroll
    for (int tc = 0; tc < 8; ++tc) {
      const bf16* bp = sXf + (((cG * 8 + tc) * 16 + l15) * 2 + hiL) * 16;
      v8bf blo = *(const v8bf*)bp;
      v8bf bhi = *(const v8bf*)(bp + 8);
      acc[tc] = wmma_bf16(a, CAT16(blo, bhi), acc[tc]);
    }
    __syncthreads();
  }

  // fused LIF epilogue
  const bool isQ = (mat == 0), isV = (mat == 2);
  const int mrow = m0 + rT * 16 + (hiL ? 8 : 0);
  #pragma unroll
  for (int tc = 0; tc < 8; ++tc) {
    const int n = n0 + (cG * 8 + tc) * 16 + l15;
    v8bf qv;
    #pragma unroll
    for (int j = 0; j < 8; ++j) {
      int o = mrow + j;
      size_t idx = ((size_t)b * C_ + o) * N_ + n;
      float y   = acc[tc][j];
      float mo  = memin[idx];
      float rst = (mo > THR_) ? THR_ : 0.0f;
      float mn  = BETA_ * mo + y - rst;
      memout[idx] = mn;
      bool spk = (mn - THR_) > 0.0f;
      if (isQ)      qv[j] = (bf16)(spk ? 2.0f : ELU0_);
      else if (isV) v_ws[idx] = (bf16)(spk ? 1.0f : -1.0f);
      else          k_ws[idx] = (bf16)(spk ? 2.0f : ELU0_);
    }
    if (isQ) { // fragment-packed q: one 16B store per tile
      int h = mrow >> 5, kh = (mrow >> 4) & 1, klb = mrow & 15;
      size_t off = (((size_t)(b * HEADS_ + h) * 256 + (n >> 4)) * 16 + (n & 15)) * 32
                   + (size_t)(kh * 16 + klb);
      *(v8bf*)(qP + off) = qv;
    }
  }
}

// ---------------------------------------------------------------------------
// Kernel 2: per (b,h) kv = k^T v (32x32, K=4096) + k_sum. grid=64, block=128.
// Async global->LDS staging; kv stored transposed bf16 (A-operand of K3).
// ---------------------------------------------------------------------------
__global__ __launch_bounds__(128)
void kv_ksum(const bf16* __restrict__ k_ws, const bf16* __restrict__ v_ws,
             bf16* __restrict__ kvT_ws, float* __restrict__ ksum_ws)
{
  const int bh = blockIdx.x;
  const int t = threadIdx.x, wave = t >> 5, lane = t & 31;
  const int mt = wave >> 1, nt = wave & 1;
  const int l15 = lane & 15, hiL = lane >> 4;

  __shared__ bf16  sK[32][72];                // 144B stride (16B aligned)
  __shared__ bf16  sV[32][72];
  __shared__ float sRed[128];

  const bf16* kb = k_ws + (size_t)bh * HD_ * N_;
  const bf16* vb = v_ws + (size_t)bh * HD_ * N_;

  v8f  acc = v8f{};
  float ks = 0.0f;

  const int sd = t >> 2, sj = (t & 3) * 16;

  for (int n0 = 0; n0 < N_; n0 += 64) {
    async2_b128((unsigned)(size_t)&sK[sd][sj], kb + (size_t)sd * N_ + n0 + sj);
    async2_b128((unsigned)(size_t)&sV[sd][sj], vb + (size_t)sd * N_ + n0 + sj);
    wait_async0();
    __syncthreads();

    { // k_sum partial
      int dd = t & 31, qq = wave * 16;
      float s = 0.0f;
      #pragma unroll
      for (int jj = 0; jj < 16; ++jj) s += (float)sK[dd][qq + jj];
      ks += s;
    }
    #pragma unroll
    for (int kk = 0; kk < 64; kk += 32) {
      const int m   = mt * 16 + l15;
      const int kbo = hiL ? 8 : 0;
      v8bf alo = *(const v8bf*)&sK[m][kk + kbo];
      v8bf ahi = *(const v8bf*)&sK[m][kk + 16 + kbo];

      const int col  = nt * 16 + l15;
      const int kbo2 = hiL ? 8 : 0;
      v8bf blo = *(const v8bf*)&sV[col][kk + kbo2];
      v8bf bhi = *(const v8bf*)&sV[col][kk + 16 + kbo2];

      acc = wmma_bf16(CAT16(alo, ahi), CAT16(blo, bhi), acc);
    }
    __syncthreads();
  }

  sRed[t] = ks;
  __syncthreads();
  if (t < 32) ksum_ws[bh * HD_ + t] = sRed[t] + sRed[t + 32] + sRed[t + 64] + sRed[t + 96];

  { // kv^T bf16
    int d2  = nt * 16 + l15;
    int d1b = mt * 16 + (hiL ? 8 : 0);
    bf16* o = kvT_ws + (size_t)bh * HD_ * HD_;
    #pragma unroll
    for (int j = 0; j < 8; ++j) o[(size_t)d2 * HD_ + d1b + j] = (bf16)acc[j];
  }
}

// ---------------------------------------------------------------------------
// Kernel 3: out^T = kv^T * q^T with SCALE/(q.k_sum+1e-6) scaling.
// q arrives fragment-packed (pure async staging); output fragment-packed.
// grid = (N/128, 64); block 256 (8 waves, 16-col tile each, 2 WMMAs).
// ---------------------------------------------------------------------------
__global__ __launch_bounds__(256)
void attn_out(const bf16* __restrict__ qP, const bf16* __restrict__ kvT_ws,
              const float* __restrict__ ksum_ws, bf16* __restrict__ attnP)
{
  const int bh  = blockIdx.y;
  const int nt0 = blockIdx.x * 8;             // 8 nt groups = 128 pixels
  const int t = threadIdx.x, wave = t >> 5, lane = t & 31;
  const int l15 = lane & 15, hiL = lane >> 4;

  __shared__ bf16  sQp[8 * 16 * 2 * 16];      // 8KB fragment-packed
  __shared__ bf16  sKV[32][48];
  __shared__ float sKS[32];

  // pure contiguous async copy of packed q strip (32B per thread)
  async2_b128((unsigned)(size_t)(sQp + t * 16),
              qP + ((size_t)bh * 256 + nt0) * 512 + (size_t)t * 16);
  if (t < 128) { // kv^T rows into padded LDS
    int row = t >> 2, seg = t & 3;
    async1_b128((unsigned)(size_t)&sKV[row][seg * 8],
                kvT_ws + (size_t)bh * 1024 + row * 32 + seg * 8);
  }
  if (t < 32) sKS[t] = ksum_ws[bh * HD_ + t];
  wait_async0();
  __syncthreads();

  const int kbo = hiL ? 8 : 0;
  v8bf a0lo = *(const v8bf*)&sKV[l15][kbo];
  v8bf a0hi = *(const v8bf*)&sKV[l15][16 + kbo];
  v8bf a1lo = *(const v8bf*)&sKV[l15 + 16][kbo];
  v8bf a1hi = *(const v8bf*)&sKV[l15 + 16][16 + kbo];

  const bf16* bp = sQp + ((wave * 16 + l15) * 2) * 16 + (hiL ? 8 : 0);
  v8bf blo = *(const v8bf*)bp;
  v8bf bhi = *(const v8bf*)(bp + 16);
  v16bf bb = CAT16(blo, bhi);

  v8f c0 = v8f{}, c1 = v8f{};
  c0 = wmma_bf16(CAT16(a0lo, a0hi), bb, c0);
  c1 = wmma_bf16(CAT16(a1lo, a1hi), bb, c1);

  // denominator: packed q column is 64B contiguous, d = kh*16+kl order
  const bf16* qq = sQp + ((wave * 16 + l15) * 2) * 16;
  float denom = 1e-6f;
  #pragma unroll
  for (int d = 0; d < 32; ++d) denom += (float)qq[d] * sKS[d];
  const float inv = SCALE_ / denom;

  v8bf o0, o1;
  #pragma unroll
  for (int j = 0; j < 8; ++j) { o0[j] = (bf16)(c0[j] * inv); o1[j] = (bf16)(c1[j] * inv); }
  const size_t baseo = (((size_t)bh * 256 + nt0 + wave) * 16 + l15) * 32;
  const int klb = hiL ? 8 : 0;
  *(v8bf*)(attnP + baseo + klb)      = o0;   // kh=0 rows
  *(v8bf*)(attnP + baseo + 16 + klb) = o1;   // kh=1 rows
}

// ---------------------------------------------------------------------------
// Kernel 4: Wo conv1x1 GEMM; X arrives fragment-packed -> pure async staging.
// grid = (N/256, C/64, B); block 256. Raw f32 output to d_out.
// ---------------------------------------------------------------------------
__global__ __launch_bounds__(256)
void wo_gemm(const bf16* __restrict__ attnP, const float* __restrict__ Wo,
             float* __restrict__ out)
{
  const int b   = blockIdx.z;
  const int m0  = blockIdx.y * 64;
  const int n0  = blockIdx.x * 256;
  const int nt0 = n0 >> 4;

  __shared__ bf16 sW[64][48];
  __shared__ bf16 sXf[16 * 16 * 2 * 16];      // 16KB fragment-packed

  const int t = threadIdx.x, wave = t >> 5, lane = t & 31;
  const int rT = wave & 3, cG = wave >> 2;
  const int l15 = lane & 15, hiL = lane >> 4;

  v8f acc[8];
  #pragma unroll
  for (int i = 0; i < 8; ++i) acc[i] = v8f{};

  const int wrow = t >> 2, wcol = (t & 3) * 8;

  for (int k0 = 0; k0 < C_; k0 += 32) {
    {
      const float4* s4 = (const float4*)(Wo + (size_t)(m0 + wrow) * C_ + k0 + wcol);
      float4 f0 = s4[0], f1 = s4[1];
      uint4 u;
      u.x = pkbf(f0.x, f0.y); u.y = pkbf(f0.z, f0.w);
      u.z = pkbf(f1.x, f1.y); u.w = pkbf(f1.z, f1.w);
      *(uint4*)&sW[wrow][wcol] = u;
    }
    // 16KB contiguous packed strip: 64B async per thread
    async4_b128((unsigned)(size_t)(sXf + t * 32),
                attnP + (((size_t)b * HEADS_ + (k0 >> 5)) * 256 + nt0) * 512
                      + (size_t)t * 32);
    wait_async0();
    __syncthreads();

    const int m  = rT * 16 + l15;
    const int kb = hiL ? 16 : 0;
    v8bf alo = *(const v8bf*)&sW[m][kb];
    v8bf ahi = *(const v8bf*)&sW[m][kb + 8];
    v16bf a = CAT16(alo, ahi);

    #pragma unroll
    for (int tc = 0; tc < 8; ++tc) {
      const bf16* bp = sXf + (((cG * 8 + tc) * 16 + l15) * 2 + hiL) * 16;
      v8bf blo = *(const v8bf*)bp;
      v8bf bhi = *(const v8bf*)(bp + 8);
      acc[tc] = wmma_bf16(a, CAT16(blo, bhi), acc[tc]);
    }
    __syncthreads();
  }

  const int mrow = m0 + rT * 16 + (hiL ? 8 : 0);
  #pragma unroll
  for (int tc = 0; tc < 8; ++tc) {
    const int n = n0 + (cG * 8 + tc) * 16 + l15;
    #pragma unroll
    for (int j = 0; j < 8; ++j)
      out[((size_t)b * C_ + (mrow + j)) * N_ + n] = acc[tc][j];
  }
}

// ---------------------------------------------------------------------------
// Kernel 5a: per-channel batch statistics (deterministic tree reduction).
// ---------------------------------------------------------------------------
__global__ __launch_bounds__(256)
void bn_stats(const float* __restrict__ y, float* __restrict__ stats)
{
  const int c = blockIdx.x, t = threadIdx.x;
  float s = 0.0f, s2 = 0.0f;
  for (int b = 0; b < B_; ++b) {
    const float* p = y + ((size_t)b * C_ + c) * N_;
    for (int n = t; n < N_; n += 256) { float v = p[n]; s += v; s2 += v * v; }
  }
  __shared__ float rs[256], rq[256];
  rs[t] = s; rq[t] = s2;
  __syncthreads();
  for (int off = 128; off; off >>= 1) {
    if (t < off) { rs[t] += rs[t + off]; rq[t] += rq[t + off]; }
    __syncthreads();
  }
  if (t == 0) {
    const float M = (float)(B_ * N_);
    float mean = rs[0] / M;
    float var  = rq[0] / M - mean * mean;
    stats[c]      = mean;
    stats[C_ + c] = rsqrtf(var + 1e-5f);
  }
}

// ---------------------------------------------------------------------------
// Kernel 5b: in-place normalize with gamma/beta.
// ---------------------------------------------------------------------------
__global__ __launch_bounds__(256)
void bn_apply(float* __restrict__ y, const float* __restrict__ stats,
              const float* __restrict__ gamma, const float* __restrict__ beta)
{
  size_t i = (size_t)blockIdx.x * 256 + threadIdx.x;
  const size_t total = (size_t)B_ * C_ * N_;
  if (i < total) {
    int c = (int)((i / N_) % C_);
    y[i] = (y[i] - stats[c]) * stats[C_ + c] * gamma[c] + beta[c];
  }
}

// ---------------------------------------------------------------------------
extern "C" void kernel_launch(void* const* d_in, const int* in_sizes, int n_in,
                              void* d_out, int out_size, void* d_ws, size_t ws_size,
                              hipStream_t stream)
{
  (void)in_sizes; (void)n_in; (void)out_size; (void)ws_size;

  const float* x     = (const float*)d_in[0];
  const float* qmem  = (const float*)d_in[1];
  const float* kmem  = (const float*)d_in[2];
  const float* vmem  = (const float*)d_in[3];
  const float* Wq    = (const float*)d_in[4];
  const float* Wk    = (const float*)d_in[5];
  const float* Wv    = (const float*)d_in[6];
  const float* Wo    = (const float*)d_in[7];
  const float* gamma = (const float*)d_in[8];
  const float* beta  = (const float*)d_in[9];
  float* out = (float*)d_out;

  const size_t BCN = (size_t)B_ * C_ * N_;        // 8,388,608
  bf16*  qP      = (bf16*)d_ws;                   // fragment-packed q
  bf16*  k_ws    = qP + BCN;                      // channel-major k
  bf16*  v_ws    = k_ws + BCN;                    // channel-major v
  bf16*  attnP   = v_ws + BCN;                    // fragment-packed attn
  bf16*  kvT_ws  = attnP + BCN;                   // 64 * 32 * 32 bf16
  float* ksum_ws = (float*)(kvT_ws + 64 * HD_ * HD_);
  float* stats   = ksum_ws + 64 * HD_;

  qkv_gemm_lif<<<dim3(N_ / 256, C_ / 64, 3 * B_), dim3(256), 0, stream>>>(
      x, Wq, Wk, Wv, qmem, kmem, vmem, out, qP, k_ws, v_ws);

  kv_ksum<<<dim3(B_ * HEADS_), dim3(128), 0, stream>>>(k_ws, v_ws, kvT_ws, ksum_ws);

  attn_out<<<dim3(N_ / 128, B_ * HEADS_), dim3(256), 0, stream>>>(
      qP, kvT_ws, ksum_ws, attnP);

  wo_gemm<<<dim3(N_ / 256, C_ / 64, B_), dim3(256), 0, stream>>>(attnP, Wo, out);

  bn_stats<<<dim3(C_), dim3(256), 0, stream>>>(out, stats);

  bn_apply<<<dim3((unsigned)((BCN + 255) / 256)), dim3(256), 0, stream>>>(
      out, stats, gamma, beta);
}
